// CLLoss_66537633349977
// MI455X (gfx1250) — matrix operand order repaired and checked by reading, not compiled, not run on hardware
//
#include <hip/hip_runtime.h>
#include <hip/hip_bf16.h>
#include <stdint.h>

// ---------------- problem constants ----------------
#define NL    1024
#define NU    3072
#define NN    4096          // NL + NU
#define NAB   8192
#define NALL  16384
#define D     1024
#define NUM_POS 16
#define INV_T 10.0f
#define THRESH 0.1f

// ---------------- WMMA types ----------------
typedef __attribute__((ext_vector_type(16))) __bf16 v16bf;
typedef __attribute__((ext_vector_type(8)))  float  v8f;
typedef __attribute__((ext_vector_type(4)))  unsigned int u32x4;

union FragBF {
    v16bf v;
    u32x4 q[2];
};

// round-to-nearest-even f32 -> bf16 (bit pattern)
static __device__ __forceinline__ unsigned short f2bf(float f) {
    unsigned int u = __float_as_uint(f);
    unsigned int r = (u + 0x7FFFu + ((u >> 16) & 1u)) >> 16;
    return (unsigned short)r;
}

// CDNA5 async copy: global -> LDS, 16B per lane, tracked by ASYNCcnt.
// ldsOff = workgroup-relative LDS byte offset (low 32 bits of generic ptr).
static __device__ __forceinline__ void async_ld16(unsigned ldsOff, const void* g) {
    asm volatile("global_load_async_to_lds_b128 %0, %1, off"
                 :: "v"(ldsOff), "v"((unsigned long long)(uintptr_t)g)
                 : "memory");
}

// ---------------- kernel 1: convert + zero accumulators ----------------
__global__ void k_init(const float* __restrict__ f_l, const float* __restrict__ f_u,
                       const float* __restrict__ f_ab,
                       unsigned short* __restrict__ Abf, unsigned short* __restrict__ Bbf,
                       float* __restrict__ zero_region /* negsum(NN)+possum(NN)+scal(8) */) {
    const int nA = NN * D;           // 4.19M
    const int nB = NAB * D;          // 8.39M
    const int nZ = 2 * NN + 8;
    const int total = nA + nB + nZ;
    for (int idx = blockIdx.x * blockDim.x + threadIdx.x; idx < total;
         idx += gridDim.x * blockDim.x) {
        if (idx < nA) {
            float v = (idx < NL * D) ? f_l[idx] : f_u[idx - NL * D];
            Abf[idx] = f2bf(v);
        } else if (idx < nA + nB) {
            Bbf[idx - nA] = f2bf(f_ab[idx - nA]);
        } else {
            zero_region[idx - nA - nB] = 0.0f;
        }
    }
}

// ---------------- kernel 2: fused bf16 WMMA GEMM + exp row-sum ----------------
// Block tile 256x128, BK=32, 8 waves as 4(M) x 2(N); wave tile 64x64 = 16 WMMAs.
// Staging: ping-pong double buffer, compile-time buffer index, async LDS copies.
#define BM 256
#define BN 128
#define BK 32
#define LDSP 40   // row stride in bf16 elems (80 B: 16B-aligned, bank-staggered)

__global__ __launch_bounds__(256) void k_gemm_exp(
        const unsigned short* __restrict__ Abf,   // [NN][D] bf16 (f_n)
        const unsigned short* __restrict__ Bbf,   // [NAB][D] bf16 (f_ab)
        float* __restrict__ negsum) {             // [NN]
    __shared__ __align__(16) unsigned short As[2][BM * LDSP];  // 2 x 20KB
    __shared__ __align__(16) unsigned short Bs[2][BN * LDSP];  // 2 x 10KB

    const int tid   = threadIdx.x;
    const int lane  = tid & 31;
    const int wid   = tid >> 5;
    const int waveM = wid & 3;        // 0..3  -> 64 rows each
    const int waveN = wid >> 2;       // 0..1  -> 64 cols each
    const int l15   = lane & 15;
    const int hgrp  = lane >> 4;      // lane group 0/1
    const int bm    = blockIdx.y * BM;
    const int bn    = blockIdx.x * BN;

    // per-thread staging addresses, computed once
    const int col8 = (tid & 3) * 8;   // chunk column (8 bf16 = 16B)
    const int r0   = tid >> 2;        // chunk row base (4 chunks per 32-elem row)
    const unsigned short* gA[4];
    const unsigned short* gB[2];
    unsigned ldsA[2][4], ldsB[2][2];
    #pragma unroll
    for (int it = 0; it < 4; ++it) {  // A tile: 256 rows, this thread covers 4
        gA[it] = Abf + (size_t)(bm + r0 + it * 64) * D + col8;
        ldsA[0][it] = (unsigned)(uintptr_t)&As[0][(r0 + it * 64) * LDSP + col8];
        ldsA[1][it] = (unsigned)(uintptr_t)&As[1][(r0 + it * 64) * LDSP + col8];
    }
    #pragma unroll
    for (int it = 0; it < 2; ++it) {  // B tile: 128 rows, this thread covers 2
        gB[it] = Bbf + (size_t)(bn + r0 + it * 64) * D + col8;
        ldsB[0][it] = (unsigned)(uintptr_t)&Bs[0][(r0 + it * 64) * LDSP + col8];
        ldsB[1][it] = (unsigned)(uintptr_t)&Bs[1][(r0 + it * 64) * LDSP + col8];
    }

    auto issueTile = [&](int buf, int kk) {       // 6 async instructions / wave
        #pragma unroll
        for (int it = 0; it < 4; ++it) async_ld16(ldsA[buf][it], gA[it] + kk);
        #pragma unroll
        for (int it = 0; it < 2; ++it) async_ld16(ldsB[buf][it], gB[it] + kk);
    };

    v8f acc[4][4] = {};

    auto compute = [&](int buf) {
        // A frags: per-lane k order {0..7,16..23} (grp0) / {8..15,24..31} (grp1)
        FragBF a[4], b[4];
        #pragma unroll
        for (int mt = 0; mt < 4; ++mt) {
            int r = waveM * 64 + mt * 16 + l15;
            const unsigned short* p = &As[buf][r * LDSP + hgrp * 8];
            a[mt].q[0] = *(const u32x4*)p;
            a[mt].q[1] = *(const u32x4*)(p + 16);
        }
        // B frags: lane = N%16, k order {0..7,16..23} (grp0) / {8..15,24..31} (grp1)
        #pragma unroll
        for (int nt = 0; nt < 4; ++nt) {
            int cdx = waveN * 64 + nt * 16 + l15;
            const unsigned short* p = &Bs[buf][cdx * LDSP + hgrp * 8];
            b[nt].q[0] = *(const u32x4*)p;
            b[nt].q[1] = *(const u32x4*)(p + 16);
        }
        #pragma unroll
        for (int mt = 0; mt < 4; ++mt)
            #pragma unroll
            for (int nt = 0; nt < 4; ++nt)
                acc[mt][nt] = __builtin_amdgcn_wmma_f32_16x16x32_bf16(
                    false, a[mt].v, false, b[nt].v,
                    (short)0, acc[mt][nt], false, false);
    };

    issueTile(0, 0);                              // prologue fill

    // D/BK = 32 tiles -> 15 straight-line ping-pong pairs + 2-tile tail
    for (int kk = 0; kk < D - 2 * BK; kk += 2 * BK) {
        issueTile(1, kk + BK);                    // fill buf1 while computing buf0
        __builtin_prefetch(gA[0] + kk + 2 * BK, 0, 0);   // warm L2 ahead
        __builtin_prefetch(gB[0] + kk + 2 * BK, 0, 0);
        asm volatile("s_wait_asynccnt 6" ::: "memory");  // tile in buf0 landed
        __syncthreads();
        compute(0);
        __syncthreads();

        issueTile(0, kk + 2 * BK);                // fill buf0 while computing buf1
        asm volatile("s_wait_asynccnt 6" ::: "memory");  // tile in buf1 landed
        __syncthreads();
        compute(1);
        __syncthreads();
    }
    // tail: tile D-2*BK is in-flight to buf0; issue final tile to buf1
    issueTile(1, D - BK);
    asm volatile("s_wait_asynccnt 6" ::: "memory");
    __syncthreads();
    compute(0);
    __syncthreads();
    asm volatile("s_wait_asynccnt 0" ::: "memory");
    __syncthreads();
    compute(1);

    // epilogue: per-row sum of exp(c/T); C layout: VGPR j -> row mt*16 + hgrp*8 + j
    #pragma unroll
    for (int mt = 0; mt < 4; ++mt) {
        #pragma unroll
        for (int j = 0; j < 8; ++j) {
            float s = 0.0f;
            #pragma unroll
            for (int nt = 0; nt < 4; ++nt)
                s += __expf(acc[mt][nt][j] * INV_T);
            #pragma unroll
            for (int m = 1; m < 16; m <<= 1)   // reduce within 16-lane N group
                s += __shfl_xor(s, m, 32);
            if (l15 == 0) {
                int rowg = bm + waveM * 64 + mt * 16 + hgrp * 8 + j;
                atomicAdd(&negsum[rowg], s);
            }
        }
    }
}

// ---------------- kernel 3: positives (16 hashed + self-noise dot) ----------------
__global__ __launch_bounds__(256) void k_pos(
        const float* __restrict__ f_l, const float* __restrict__ f_u,
        const float* __restrict__ f_ln, const float* __restrict__ f_un,
        float* __restrict__ possum) {
    const int i    = blockIdx.x;
    const int lane = threadIdx.x & 31;
    const int wid  = threadIdx.x >> 5;
    const float* a = (i < NL) ? (f_l + (size_t)i * D) : (f_u + (size_t)(i - NL) * D);

    for (int p = wid; p < NUM_POS + 1; p += 8) {
        const float* bptr;
        if (p == NUM_POS) {
            bptr = (i < NL) ? (f_ln + (size_t)i * D) : (f_un + (size_t)(i - NL) * D);
        } else {
            // deterministic splitmix-style draw over [0, NN-1) with self-skip
            unsigned long long h = (unsigned long long)(i + 1) * 0x9E3779B97F4A7C15ull
                                 ^ (unsigned long long)(p + 1) * 0xBF58476D1CE4E5B9ull;
            h ^= h >> 30; h *= 0x94D049BB133111EBull; h ^= h >> 31;
            unsigned int j = (unsigned int)(h % (unsigned long long)(NN - 1));
            if ((int)j >= i) j++;
            bptr = (j < NL) ? (f_l + (size_t)j * D) : (f_u + (size_t)(j - NL) * D);
        }
        float s = 0.0f;
        for (int k = lane; k < D; k += 32) s += a[k] * bptr[k];
        #pragma unroll
        for (int m = 1; m < 32; m <<= 1) s += __shfl_xor(s, m, 32);
        if (lane == 0) atomicAdd(&possum[i], __expf(s * INV_T));
    }
}

// ---------------- kernel 4: terms 0..2 ----------------
__global__ void k_terms(const float* __restrict__ y, const float* __restrict__ batch_y,
                        const float* __restrict__ y_l, const float* __restrict__ y_u,
                        const float* __restrict__ y_un, const int* __restrict__ u_near,
                        float* __restrict__ scal) {
    float t0 = 0.f, t1 = 0.f, t2 = 0.f, c2 = 0.f;
    for (int idx = blockIdx.x * blockDim.x + threadIdx.x; idx < NU;
         idx += gridDim.x * blockDim.x) {
        if (idx < NL) { float d = batch_y[idx] - y_l[idx]; t0 += d * d; }
        float d1 = y_u[idx] - y_un[idx]; t1 += d1 * d1;
        int un = u_near[idx];
        if (un != -1) {
            c2 += 1.0f;
            float dd = fabsf(y_u[idx] - y[un]);
            if (dd >= THRESH) t2 += dd * dd;
        }
    }
    #pragma unroll
    for (int m = 1; m < 32; m <<= 1) {
        t0 += __shfl_xor(t0, m, 32); t1 += __shfl_xor(t1, m, 32);
        t2 += __shfl_xor(t2, m, 32); c2 += __shfl_xor(c2, m, 32);
    }
    if ((threadIdx.x & 31) == 0) {
        atomicAdd(&scal[0], t0); atomicAdd(&scal[1], t1);
        atomicAdd(&scal[2], t2); atomicAdd(&scal[3], c2);
    }
}

// ---------------- kernel 5: finalize ----------------
__global__ void k_final(const float* __restrict__ possum, const float* __restrict__ negsum,
                        const float* __restrict__ scal, float* __restrict__ out) {
    float s = 0.0f;
    for (int i = threadIdx.x; i < NN; i += blockDim.x) {
        float ps = possum[i];
        s += logf(ps / (ps + negsum[i]));
    }
    #pragma unroll
    for (int m = 1; m < 32; m <<= 1) s += __shfl_xor(s, m, 32);
    __shared__ float w[8];
    if ((threadIdx.x & 31) == 0) w[threadIdx.x >> 5] = s;
    __syncthreads();
    if (threadIdx.x == 0) {
        float tot = 0.f;
        #pragma unroll
        for (int k = 0; k < 8; ++k) tot += w[k];
        float term3 = -tot / (float)NN;
        float term0 = scal[0] / (float)NL;
        float term1 = scal[1] / (float)NU;
        float cnt   = scal[3] < 1.0f ? 1.0f : scal[3];
        float term2 = scal[2] / cnt;
        out[0] = term0 + term1 + term2 + term3;
        out[1] = term0;
        out[2] = term1;   // ALPHA[0] = 1
        out[3] = term2;   // ALPHA[1] = 1
        out[4] = term3;   // ALPHA[2] = 1
    }
}

// ---------------- launch ----------------
extern "C" void kernel_launch(void* const* d_in, const int* in_sizes, int n_in,
                              void* d_out, int out_size, void* d_ws, size_t ws_size,
                              hipStream_t stream) {
    const float* f_l     = (const float*)d_in[0];
    const float* f_ln    = (const float*)d_in[1];
    const float* f_u     = (const float*)d_in[2];
    const float* f_un    = (const float*)d_in[3];
    const float* f_ab    = (const float*)d_in[4];
    const float* y       = (const float*)d_in[5];
    const float* batch_y = (const float*)d_in[6];
    const float* y_l     = (const float*)d_in[7];
    const float* y_u     = (const float*)d_in[8];
    const float* y_un    = (const float*)d_in[9];
    const int*   u_near  = (const int*)d_in[10];
    float* out = (float*)d_out;

    // workspace layout
    unsigned short* Abf = (unsigned short*)d_ws;               // NN*D bf16   (8 MB)
    unsigned short* Bbf = Abf + (size_t)NN * D;                // NAB*D bf16 (16 MB)
    float* negsum = (float*)(Bbf + (size_t)NAB * D);           // NN
    float* possum = negsum + NN;                               // NN
    float* scal   = possum + NN;                               // 8

    k_init<<<2048, 256, 0, stream>>>(f_l, f_u, f_ab, Abf, Bbf, negsum);
    k_gemm_exp<<<dim3(NAB / BN, NN / BM), 256, 0, stream>>>(Abf, Bbf, negsum);
    k_pos<<<NN, 256, 0, stream>>>(f_l, f_u, f_ln, f_un, possum);
    k_terms<<<16, 256, 0, stream>>>(y, batch_y, y_l, y_u, y_un, u_near, scal);
    k_final<<<1, 256, 0, stream>>>(possum, negsum, scal, out);
}
